// RelativeMultiHeadAttention_37572373905989
// MI455X (gfx1250) — matrix operand order repaired and checked by reading, not compiled, not run on hardware
//
#include <hip/hip_runtime.h>

// ---------------------------------------------------------------------------
// RelativeMultiHeadAttention for MI455X (gfx1250), bf16 WMMA throughout.
// B=2, S=2048, D_MODEL=2048, H=16, D_HEAD=128, N_BUCKETS=64
// ---------------------------------------------------------------------------

#define B_   2
#define S_   2048
#define DM_  2048
#define H_   16
#define DH_  128

typedef __attribute__((ext_vector_type(16))) __bf16 v16bf;
typedef __attribute__((ext_vector_type(8)))  float  v8f;

union Frag { uint4 u[2]; v16bf v; };

__device__ __forceinline__ unsigned short f2bf(float f) {
    union { float f; unsigned int u; } v; v.f = f;
    unsigned int r = v.u + 0x7FFFu + ((v.u >> 16) & 1u);   // RNE
    return (unsigned short)(r >> 16);
}
__device__ __forceinline__ float bf2f(unsigned short h) {
    union { unsigned int u; float f; } v; v.u = ((unsigned int)h) << 16;
    return v.f;
}

// A-fragment (16x32 bf16). p points at (row0,k0); ld = row stride in shorts.
// lanes 0-15: row=lane,    K = {0..7, 16..23}
// lanes16-31: row=lane-16, K = {8..15, 24..31}
__device__ __forceinline__ v16bf load_frag_A(const unsigned short* p, int ld) {
    int lane = threadIdx.x & 31;
    int half = lane >> 4, lr = lane & 15;
    const unsigned short* row = p + (size_t)lr * ld + half * 8;
    Frag f;
    f.u[0] = *(const uint4*)(row);
    f.u[1] = *(const uint4*)(row + 16);
    return f.v;
}

// B-fragment (32x16 bf16). Source stored "transposed": each B-column is a
// contiguous run of 32 K-values. p points at (col0,k0); ld = col stride.
// lanes 0-15: col=lane,    K = {0..7, 16..23} ; lanes 16-31: col=lane-16, K = {8..15, 24..31}
__device__ __forceinline__ v16bf load_frag_B(const unsigned short* p, int ld) {
    int lane = threadIdx.x & 31;
    int half = lane >> 4, lr = lane & 15;
    const unsigned short* row = p + (size_t)lr * ld + half * 8;
    Frag f;
    f.u[0] = *(const uint4*)(row);
    f.u[1] = *(const uint4*)(row + 16);
    return f.v;
}

__device__ __forceinline__ v8f wmma_bf(v16bf a, v16bf b, v8f c) {
    return __builtin_amdgcn_wmma_f32_16x16x32_bf16(false, a, false, b,
                                                   (short)0, c, false, false);
}

// ---------------------------------------------------------------------------
// K1a: fp32 -> bf16 raw convert
__global__ void cvt_bf(const float* __restrict__ in, unsigned short* __restrict__ out, int n) {
    int i = blockIdx.x * blockDim.x + threadIdx.x;
    if (i < n) out[i] = f2bf(in[i]);
}

// K1b: transpose+convert weight [K=2048][N] -> [N][2048] bf16
__global__ void cvtT_bf(const float* __restrict__ in, unsigned short* __restrict__ out, int N) {
    int i = blockIdx.x * blockDim.x + threadIdx.x;
    if (i >= N * DM_) return;
    int kk = i & (DM_ - 1);
    int n  = i >> 11;
    out[i] = f2bf(in[(size_t)kk * N + n]);
}

// K3: gather bucket embeddings -> bf16
__global__ void gather_rel(const int* __restrict__ rrb, const int* __restrict__ rwb,
                           const float* __restrict__ remb,
                           unsigned short* __restrict__ rre, unsigned short* __restrict__ rwe) {
    int i = blockIdx.x * blockDim.x + threadIdx.x;       // B*S*DH
    if (i >= B_ * S_ * DH_) return;
    int d = i & (DH_ - 1);
    int bs = i >> 7;
    rre[i] = f2bf(remb[rrb[bs] * DH_ + d]);
    rwe[i] = f2bf(remb[rwb[bs] * DH_ + d]);
}

// K4: alpha[b,h,s] = q . r_r_bias[h] ; beta[b,h,s] = k . r_w_bias[h]
__global__ void bias_dot(const unsigned short* __restrict__ qbf, const unsigned short* __restrict__ kbf,
                         const float* __restrict__ rrb, const float* __restrict__ rwb,
                         float* __restrict__ alpha, float* __restrict__ beta) {
    int i = blockIdx.x * blockDim.x + threadIdx.x;       // 2 * B*H*S
    int total = B_ * H_ * S_;
    if (i >= 2 * total) return;
    int which = (i >= total);
    int j = which ? (i - total) : i;                     // (b*H+h)*S + s
    int h = (j >> 11) & (H_ - 1);
    const unsigned short* src = (which ? kbf : qbf) + (size_t)j * DH_;
    const float* w = (which ? rwb : rrb) + h * DH_;
    float sum = 0.f;
    for (int d = 0; d < DH_; ++d) sum += bf2f(src[d]) * w[d];
    (which ? beta : alpha)[j] = sum;
}

// ---------------------------------------------------------------------------
// K2/K6: bf16 WMMA GEMM, C[M,N] = A[M,2048] * Bt[N,2048]^T, M=4096.
// 256 threads = 8 waves; 128x128 tile; wave = 32x64 (2x4 grid of 16x16 WMMA).
// mode 0: scatter fp32->bf16 into q / k head-major and v transposed.
// mode 1: fp32 row-major into outf[M][N].
__global__ void __launch_bounds__(256, 1)
gemm_bf16(const unsigned short* __restrict__ A, const unsigned short* __restrict__ Bt,
          int N, int mode,
          unsigned short* __restrict__ qd, unsigned short* __restrict__ kd,
          unsigned short* __restrict__ vtd, float* __restrict__ outf) {
    __shared__ unsigned short sA[128 * 32];
    __shared__ unsigned short sB[128 * 32];

    const int t = threadIdx.x;
    const int tileM = blockIdx.x * 128;
    const int tileN = blockIdx.y * 128;
    const int wid = t >> 5;
    const int wm = (wid & 3) * 32;
    const int wn = (wid >> 2) * 64;

    v8f acc[2][4];
    for (int i = 0; i < 2; ++i)
        for (int j = 0; j < 4; ++j)
            acc[i][j] = (v8f){0.f,0.f,0.f,0.f,0.f,0.f,0.f,0.f};

    for (int kt = 0; kt < DM_ / 32; ++kt) {
        // stage 128x32 tiles of A and Bt into LDS (uint4 = 8 bf16)
        for (int i = 0; i < 2; ++i) {
            int idx = t + i * 256;              // 0..511
            int row = idx >> 2;
            int quad = idx & 3;
            *(uint4*)(sA + row * 32 + quad * 8) =
                *(const uint4*)(A + (size_t)(tileM + row) * DM_ + kt * 32 + quad * 8);
            *(uint4*)(sB + row * 32 + quad * 8) =
                *(const uint4*)(Bt + (size_t)(tileN + row) * DM_ + kt * 32 + quad * 8);
        }
        __syncthreads();

        v16bf a0 = load_frag_A(sA + wm * 32, 32);
        v16bf a1 = load_frag_A(sA + (wm + 16) * 32, 32);
        for (int j = 0; j < 4; ++j) {
            v16bf bf = load_frag_B(sB + (wn + j * 16) * 32, 32);
            acc[0][j] = wmma_bf(a0, bf, acc[0][j]);
            acc[1][j] = wmma_bf(a1, bf, acc[1][j]);
        }
        __syncthreads();
    }

    const int lane = t & 31, half = lane >> 4, lr = lane & 15;
    for (int i = 0; i < 2; ++i)
        for (int j = 0; j < 4; ++j)
            for (int g = 0; g < 8; ++g) {
                int rowl = wm + i * 16 + half * 8 + g;
                int coll = wn + j * 16 + lr;
                size_t m = (size_t)tileM + rowl;     // b*S + s
                int n = tileN + coll;
                float val = acc[i][j][g];
                if (mode == 1) {
                    outf[m * (size_t)N + n] = val;
                } else {
                    int which = n >> 11;             // 0=q 1=k 2=v
                    int c = n & (DM_ - 1);
                    int h = c >> 7;
                    int d = c & (DH_ - 1);
                    int b = (int)(m >> 11);
                    int s = (int)(m & (S_ - 1));
                    if (which == 0)
                        qd[(((size_t)(b * H_ + h)) * S_ + s) * DH_ + d] = f2bf(val);
                    else if (which == 1)
                        kd[(((size_t)(b * H_ + h)) * S_ + s) * DH_ + d] = f2bf(val);
                    else
                        vtd[(((size_t)(b * H_ + h)) * DH_ + d) * S_ + s] = f2bf(val);
                }
            }
}

// ---------------------------------------------------------------------------
// K5: flash attention. One wave per (b, h, 16-query block). 8 waves/block.
// scores = Q.K^T + diag(alpha).(RRE.K^T) + (Q.RWE^T).diag(beta) + causal
__global__ void __launch_bounds__(256, 1)
attn_kernel(const unsigned short* __restrict__ q, const unsigned short* __restrict__ k,
            const unsigned short* __restrict__ vT,
            const unsigned short* __restrict__ rre, const unsigned short* __restrict__ rwe,
            const float* __restrict__ alpha, const float* __restrict__ beta,
            unsigned short* __restrict__ obf) {
    __shared__ unsigned short pshare[8][16 * 32];        // per-wave P staging

    const int wid_g = blockIdx.x * 8 + (threadIdx.x >> 5);
    const int qblk = wid_g & (S_ / 16 - 1);
    const int h = (wid_g >> 7) & (H_ - 1);
    const int b = wid_g >> 11;
    const int x0 = qblk * 16;
    const int lane = threadIdx.x & 31, half = lane >> 4, lr = lane & 15;

    const unsigned short* qbase = q + (((size_t)(b * H_ + h)) * S_ + x0) * DH_;
    const unsigned short* kb   = k  + ((size_t)(b * H_ + h)) * S_ * DH_;
    const unsigned short* vtb  = vT + ((size_t)(b * H_ + h)) * DH_ * S_;
    const unsigned short* rreb = rre + ((size_t)b * S_ + x0) * DH_;
    const unsigned short* rweb = rwe + ((size_t)b * S_) * DH_;
    const float* alp = alpha + ((size_t)(b * H_ + h)) * S_ + x0;
    const float* bet = beta  + ((size_t)(b * H_ + h)) * S_;

    v16bf qf[4], rf[4];
    for (int c = 0; c < 4; ++c) {
        qf[c] = load_frag_A(qbase + c * 32, DH_);
        rf[c] = load_frag_A(rreb + c * 32, DH_);
    }
    float alphaR[8];
    for (int g = 0; g < 8; ++g) alphaR[g] = alp[half * 8 + g];

    float mrow[8], lrow[8];
    v8f oacc[8];
    for (int g = 0; g < 8; ++g) { mrow[g] = -1e30f; lrow[g] = 0.f; }
    for (int ch = 0; ch < 8; ++ch)
        oacc[ch] = (v8f){0.f,0.f,0.f,0.f,0.f,0.f,0.f,0.f};

    unsigned short* pl = &pshare[threadIdx.x >> 5][0];
    const int ylim = x0 + 15;                            // causal: keys <= query

    for (int y0 = 0; y0 <= ylim; y0 += 32) {
        v8f sQa = (v8f){0,0,0,0,0,0,0,0}, sQb = sQa;
        v8f sRa = sQa, sRb = sQa, sWa = sQa, sWb = sQa;
        for (int c = 0; c < 4; ++c) {
            v16bf kfa = load_frag_B(kb + ((size_t)y0) * DH_ + c * 32, DH_);
            v16bf kfb = load_frag_B(kb + ((size_t)(y0 + 16)) * DH_ + c * 32, DH_);
            v16bf wfa = load_frag_B(rweb + ((size_t)y0) * DH_ + c * 32, DH_);
            v16bf wfb = load_frag_B(rweb + ((size_t)(y0 + 16)) * DH_ + c * 32, DH_);
            sQa = wmma_bf(qf[c], kfa, sQa);  sQb = wmma_bf(qf[c], kfb, sQb);
            sRa = wmma_bf(rf[c], kfa, sRa);  sRb = wmma_bf(rf[c], kfb, sRb);
            sWa = wmma_bf(qf[c], wfa, sWa);  sWb = wmma_bf(qf[c], wfb, sWb);
        }
        float betaA = bet[y0 + lr];
        float betaB = bet[y0 + 16 + lr];

        float sA[8], sB[8], corr[8];
        for (int g = 0; g < 8; ++g) {
            int row = x0 + half * 8 + g;
            int ca = y0 + lr, cb = y0 + 16 + lr;
            sA[g] = sQa[g] + alphaR[g] * sRa[g] + betaA * sWa[g] + (ca > row ? -1e9f : 0.f);
            sB[g] = sQb[g] + alphaR[g] * sRb[g] + betaB * sWb[g] + (cb > row ? -1e9f : 0.f);
        }
        // online softmax per row (rows live in 16-lane halves of C layout)
        for (int g = 0; g < 8; ++g) {
            float mt = fmaxf(sA[g], sB[g]);
            for (int o = 8; o >= 1; o >>= 1) mt = fmaxf(mt, __shfl_xor(mt, o, 32));
            float mnew = fmaxf(mrow[g], mt);
            corr[g] = __expf(mrow[g] - mnew);
            mrow[g] = mnew;
            float pa = __expf(sA[g] - mnew);
            float pb = __expf(sB[g] - mnew);
            sA[g] = pa; sB[g] = pb;
            float rs = pa + pb;
            for (int o = 8; o >= 1; o >>= 1) rs += __shfl_xor(rs, o, 32);
            lrow[g] = lrow[g] * corr[g] + rs;
        }
        for (int ch = 0; ch < 8; ++ch)
            for (int g = 0; g < 8; ++g)
                oacc[ch][g] *= corr[g];

        // P: C-layout -> A-layout via per-wave LDS staging
        for (int g = 0; g < 8; ++g) {
            pl[(half * 8 + g) * 32 + lr]      = f2bf(sA[g]);
            pl[(half * 8 + g) * 32 + 16 + lr] = f2bf(sB[g]);
        }
        asm volatile("s_wait_dscnt 0" ::: "memory");
        v16bf pf = load_frag_A(pl, 32);
        for (int ch = 0; ch < 8; ++ch) {
            v16bf vf = load_frag_B(vtb + ((size_t)(ch * 16)) * S_ + y0, S_);
            oacc[ch] = wmma_bf(pf, vf, oacc[ch]);
        }
        asm volatile("s_wait_dscnt 0" ::: "memory");
    }

    // epilogue: O /= l, write bf16 into [b][s][h*128+d] for final projection
    for (int ch = 0; ch < 8; ++ch)
        for (int g = 0; g < 8; ++g) {
            float val = oacc[ch][g] / lrow[g];
            int srow = x0 + half * 8 + g;
            int col = h * DH_ + ch * 16 + lr;
            obf[((size_t)b * S_ + srow) * DM_ + col] = f2bf(val);
        }
}

// ---------------------------------------------------------------------------
extern "C" void kernel_launch(void* const* d_in, const int* in_sizes, int n_in,
                              void* d_out, int out_size, void* d_ws, size_t ws_size,
                              hipStream_t stream) {
    const float* x      = (const float*)d_in[0];
    const int*   rrbk   = (const int*)d_in[1];
    const int*   rwbk   = (const int*)d_in[2];
    /* d_in[3] = attn_mask (causal, recomputed in-kernel) */
    const float* wqkv   = (const float*)d_in[4];
    const float* wfc    = (const float*)d_in[5];
    const float* rrbias = (const float*)d_in[6];
    const float* rwbias = (const float*)d_in[7];
    const float* remb   = (const float*)d_in[8];
    float* out = (float*)d_out;

    char* ws = (char*)d_ws;
    size_t off = 0;
    auto alloc = [&](size_t bytes) -> void* {
        void* p = ws + off;
        off += (bytes + 255) & ~(size_t)255;
        return p;
    };
    const size_t NX = (size_t)B_ * S_ * DM_;             // 8388608
    unsigned short* x_bf   = (unsigned short*)alloc(NX * 2);
    unsigned short* wqkvT  = (unsigned short*)alloc((size_t)3 * DM_ * DM_ * 2);
    unsigned short* wfcT   = (unsigned short*)alloc((size_t)DM_ * DM_ * 2);
    unsigned short* qbf    = (unsigned short*)alloc(NX * 2);
    unsigned short* kbf    = (unsigned short*)alloc(NX * 2);
    unsigned short* vtbf   = (unsigned short*)alloc(NX * 2);
    unsigned short* rrebf  = (unsigned short*)alloc((size_t)B_ * S_ * DH_ * 2);
    unsigned short* rwebf  = (unsigned short*)alloc((size_t)B_ * S_ * DH_ * 2);
    float*          alphap = (float*)alloc((size_t)B_ * H_ * S_ * 4);
    float*          betap  = (float*)alloc((size_t)B_ * H_ * S_ * 4);
    unsigned short* obf    = (unsigned short*)alloc(NX * 2);

    // 1) precision conversion + weight transposes
    cvt_bf<<<(int)(NX / 256), 256, 0, stream>>>(x, x_bf, (int)NX);
    cvtT_bf<<<(3 * DM_ * DM_) / 256, 256, 0, stream>>>(wqkv, wqkvT, 3 * DM_);
    cvtT_bf<<<(DM_ * DM_) / 256, 256, 0, stream>>>(wfc, wfcT, DM_);
    gather_rel<<<(B_ * S_ * DH_) / 256, 256, 0, stream>>>(rrbk, rwbk, remb, rrebf, rwebf);

    // 2) QKV projection (WMMA), scatter q/k head-major, v transposed
    gemm_bf16<<<dim3((B_ * S_) / 128, (3 * DM_) / 128), 256, 0, stream>>>(
        x_bf, wqkvT, 3 * DM_, 0, qbf, kbf, vtbf, nullptr);

    // 3) relative-bias scalars
    bias_dot<<<(2 * B_ * H_ * S_) / 256, 256, 0, stream>>>(
        qbf, kbf, rrbias, rwbias, alphap, betap);

    // 4) fused flash attention (WMMA scores + WMMA PV)
    attn_kernel<<<(B_ * H_ * (S_ / 16)) / 8, 256, 0, stream>>>(
        qbf, kbf, vtbf, rrebf, rwebf, alphap, betap, obf);

    // 5) output projection (WMMA), fp32 result
    gemm_bf16<<<dim3((B_ * S_) / 128, DM_ / 128), 256, 0, stream>>>(
        obf, wfcT, DM_, 1, nullptr, nullptr, nullptr, out);
}
